// MoE_11424613007415
// MI455X (gfx1250) — matrix operand order repaired and checked by reading, not compiled, not run on hardware
//
#include <hip/hip_runtime.h>

typedef unsigned short u16;
typedef unsigned int   u32;

typedef __attribute__((ext_vector_type(16))) __bf16 v16bf;
typedef __attribute__((ext_vector_type(8)))  float  v8f;

#define BB    8
#define CCH   128
#define EE    4
#define HSZ   96
#define HWSZ  (HSZ*HSZ)                 // 9216
#define PLANE ((size_t)BB*CCH*HWSZ)     // 9,437,184 elements per expert output
#define HIDLK 512
#define HIDST 256

union BF16Frag { v16bf v; u32 u[8]; };

// ---------------- small helpers ----------------
__device__ __forceinline__ u32 f2bf1(float f) {
  u32 u = __float_as_uint(f);
  return (u + 0x7FFFu + ((u >> 16) & 1u)) >> 16;   // RNE f32 -> bf16
}
__device__ __forceinline__ u32 f2bf_pk(float lo, float hi) {
  return (f2bf1(lo) & 0xFFFFu) | (f2bf1(hi) << 16);
}
__device__ __forceinline__ float bf2f(u16 v) { return __uint_as_float(((u32)v) << 16); }
__device__ __forceinline__ float gelu_exact(float x) {
  return 0.5f * x * (1.0f + erff(x * 0.70710678118654752440f));
}

// A-fragment (16x32 bf16) from row-major weights. wrow already points at
// W + (mbase + (lane&15))*ldK + kbase. Lanes 0-15 hold K 0..15 (2v pairs),
// lanes 16-31 hold K 16..31 (16+2v pairs), packed pairs per VGPR.
__device__ __forceinline__ BF16Frag load_afrag(const u16* __restrict__ wrow, int h) {
  BF16Frag A;
#pragma unroll
  for (int v = 0; v < 8; ++v) {
    int kp = h * 16 + 2 * v;
    A.u[v] = *(const u32*)(wrow + kp);
  }
  return A;
}

// Convert two 16-row f32 accumulator tiles (one 32-wide K-block of the hidden
// activations, D-layout: lane holds rows v+8h of its pixel column) into a
// B-fragment (32x16 bf16, K = h*16+2v pairs) using a half-wave shfl_xor swap.
__device__ __forceinline__ void pack_pair(const float (&hv)[2][8], int h, BF16Frag &out) {
  float ox[2][8];
#pragma unroll
  for (int ti = 0; ti < 2; ++ti)
#pragma unroll
    for (int v = 0; v < 8; ++v) ox[ti][v] = __shfl_xor(hv[ti][v], 16);
#pragma unroll
  for (int v = 0; v < 8; ++v) {
    const int r = (v & 3) * 2;                 // register index of needed row pair
    bool own = ((v >= 4) == (h != 0));         // rows 8..15 live in upper half-lanes
    float lo0 = own ? hv[0][r]     : ox[0][r];
    float hi0 = own ? hv[0][r + 1] : ox[0][r + 1];
    float lo1 = own ? hv[1][r]     : ox[1][r];
    float hi1 = own ? hv[1][r + 1] : ox[1][r + 1];
    float lo = h ? lo1 : lo0;
    float hi = h ? hi1 : hi0;
    out.u[v] = f2bf_pk(lo, hi);
  }
}

// ---------------- prep kernels ----------------
__global__ void k_cvt_bf16(const float* __restrict__ s, u16* __restrict__ d, int n) {
  int i = blockIdx.x * 256 + threadIdx.x;
  if (i < n) d[i] = (u16)f2bf1(s[i]);
}

__global__ void k_fold_bn(const float* __restrict__ g, const float* __restrict__ b,
                          const float* __restrict__ m, const float* __restrict__ v,
                          float* __restrict__ sc, float* __restrict__ sh, int n) {
  int i = blockIdx.x * 64 + threadIdx.x;
  if (i < n) {
    float s = g[i] / sqrtf(v[i] + 1e-5f);
    sc[i] = s;
    sh[i] = b[i] - m[i] * s;
  }
}

// ---------------- gating ----------------
__global__ __launch_bounds__(256) void k_gap(const float* __restrict__ x, float* __restrict__ gap) {
  __shared__ float red[256];
  int bc = blockIdx.x;                         // b*C + c
  const float4* p4 = (const float4*)(x + (size_t)bc * HWSZ);
  float s = 0.f;
  for (int i = threadIdx.x; i < HWSZ / 4; i += 256) {
    float4 v = p4[i];
    s += (v.x + v.y) + (v.z + v.w);
  }
  red[threadIdx.x] = s; __syncthreads();
  for (int o = 128; o > 0; o >>= 1) {
    if (threadIdx.x < o) red[threadIdx.x] += red[threadIdx.x + o];
    __syncthreads();
  }
  if (threadIdx.x == 0) gap[bc] = red[0] * (1.0f / (float)HWSZ);
}

__global__ __launch_bounds__(128) void k_gate(const float* __restrict__ gap,
                                              const float* __restrict__ gates,
                                              float* __restrict__ wc,
                                              float* __restrict__ loss_out) {
  __shared__ float lg[128];
  __shared__ float probs[128];
  int tid = threadIdx.x;                        // 128 = 4g * 8b * 4e
  int g = tid >> 5, b = (tid >> 2) & 7, e = tid & 3;
  float acc = 0.f;
  for (int c = 0; c < CCH; ++c) acc += gap[b * CCH + c] * gates[(g * CCH + c) * EE + e];
  lg[tid] = acc;
  __syncthreads();
  if (tid < 32) {                               // one thread per (g,b)
    const float* l = &lg[tid * 4];
    float mx = fmaxf(fmaxf(l[0], l[1]), fmaxf(l[2], l[3]));
    float p[4]; float s = 0.f;
    for (int k = 0; k < 4; ++k) { p[k] = expf(l[k] - mx); s += p[k]; }
    for (int k = 0; k < 4; ++k) { p[k] /= s; probs[tid * 4 + k] = p[k]; }
    int i1 = 0; for (int k = 1; k < 4; ++k) if (p[k] > p[i1]) i1 = k;        // first-occurrence argmax
    int i2 = (i1 == 0) ? 1 : 0;
    for (int k = 0; k < 4; ++k) if (k != i1 && p[k] > p[i2]) i2 = k;
    float den = p[i1] + p[i2] + 1e-10f;
    float* wp = &wc[tid * 4];
    wp[0] = wp[1] = wp[2] = wp[3] = 0.f;
    wp[i1] = p[i1] / den; wp[i2] = p[i2] / den;
  }
  __syncthreads();
  if (tid == 0) {
    float u[4] = {0, 0, 0, 0};
    for (int i = 0; i < 32; ++i)
      for (int k = 0; k < 4; ++k) u[k] += probs[i * 4 + k];
    float mean = 0.f;
    for (int k = 0; k < 4; ++k) { u[k] *= (1.0f / 32.0f); mean += u[k]; }
    mean *= 0.25f;
    float var = 0.f;
    for (int k = 0; k < 4; ++k) { float d = u[k] - mean; var += d * d; }
    var *= (1.0f / 3.0f);                       // ddof=1
    *loss_out = var / (mean * mean + 1e-10f);
  }
}

// ---------------- depthwise 7x7 + GELU + BN + residual (t in bf16) ----------------
__global__ __launch_bounds__(256) void k_dw(const float* __restrict__ x,
                                            const float* __restrict__ dww,
                                            const float* __restrict__ s1,
                                            const float* __restrict__ t1,
                                            u16* __restrict__ tout) {
  __shared__ float plane[HWSZ];                 // 36 KB, full (b,c) plane in LDS
  __shared__ float wl[49];
  int bc = blockIdx.x;                          // b*C + c
  int c = bc & (CCH - 1);
  const float4* xp4 = (const float4*)(x + (size_t)bc * HWSZ);
  for (int i = threadIdx.x; i < HWSZ / 4; i += 256) ((float4*)plane)[i] = xp4[i];
  if (threadIdx.x < 49) wl[threadIdx.x] = dww[c * 49 + threadIdx.x];
  __syncthreads();
  float sc = s1[c], sh = t1[c];
  u16* tp = tout + (size_t)bc * HWSZ;
  for (int i = threadIdx.x; i < HWSZ; i += 256) {
    int hh = i / HSZ, ww = i - hh * HSZ;
    float acc = 0.f;
#pragma unroll
    for (int dy = -3; dy <= 3; ++dy) {
      int y = hh + dy;
      if ((unsigned)y < (unsigned)HSZ) {
#pragma unroll
        for (int dx = -3; dx <= 3; ++dx) {
          int xw = ww + dx;
          if ((unsigned)xw < (unsigned)HSZ)
            acc += wl[(dy + 3) * 7 + (dx + 3)] * plane[y * HSZ + xw];
        }
      }
    }
    float val = plane[i] + (gelu_exact(acc) * sc + sh);
    tp[i] = (u16)f2bf1(val);
  }
}

// ---------------- LK expert fused MLP: eo = bn3(W2 @ bn2(gelu(W1 @ t))) ----------------
// 128 pixels / block, 8 waves, wave w owns pixel columns [16w,16w+16).
// The 512-wide hidden activations never leave registers.
__global__ __launch_bounds__(256) void k_lk_mlp(const u16* __restrict__ t,    // [B][C][HW] bf16
                                                const u16* __restrict__ w1,   // [512][128] bf16
                                                const u16* __restrict__ w2,   // [128][512] bf16
                                                const float* __restrict__ s2, const float* __restrict__ t2,
                                                const float* __restrict__ s3, const float* __restrict__ t3,
                                                u16* __restrict__ eo) {       // [B][C][HW] bf16
  const int lane = threadIdx.x & 31;
  const int wave = threadIdx.x >> 5;
  const int h = lane >> 4, n = lane & 15;
  const size_t pix0 = (size_t)blockIdx.x * 128 + wave * 16;   // never straddles b (9216 % 128 == 0)
  const int b = (int)(pix0 / HWSZ);
  const int hw = (int)(pix0 % HWSZ) + n;
  const u16* tb = t + (size_t)b * CCH * HWSZ + hw;

  // Preload B-fragments of t for K = 0..127 (transposed gather from channel-major t).
  BF16Frag B1[4];
#pragma unroll
  for (int kk = 0; kk < 4; ++kk) {
#pragma unroll
    for (int v = 0; v < 8; ++v) {
      int k = kk * 32 + h * 16 + 2 * v;
      u32 lo = tb[(size_t)k * HWSZ];
      u32 hi = tb[(size_t)(k + 1) * HWSZ];
      B1[kk].u[v] = (lo & 0xFFFFu) | (hi << 16);
    }
  }

  v8f oacc[8];
#pragma unroll
  for (int i = 0; i < 8; ++i) oacc[i] = (v8f){0, 0, 0, 0, 0, 0, 0, 0};

  for (int kb = 0; kb < 16; ++kb) {             // hidden channels, 32 at a time
    float hv[2][8];
#pragma unroll
    for (int ti = 0; ti < 2; ++ti) {
      v8f acc = (v8f){0, 0, 0, 0, 0, 0, 0, 0};
      const u16* wrow = w1 + (size_t)(kb * 32 + ti * 16 + n) * CCH;
      // preload all 4 A-fragments, then issue the WMMA chain (hides L2 latency)
      BF16Frag A[4];
#pragma unroll
      for (int kk = 0; kk < 4; ++kk) A[kk] = load_afrag(wrow + kk * 32, h);
#pragma unroll
      for (int kk = 0; kk < 4; ++kk)
        acc = __builtin_amdgcn_wmma_f32_16x16x32_bf16(false, A[kk].v, false, B1[kk].v,
                                                      (short)0, acc, false, false);
#pragma unroll
      for (int v = 0; v < 8; ++v) {
        int m = kb * 32 + ti * 16 + v + 8 * h;
        hv[ti][v] = gelu_exact(acc[v]) * s2[m] + t2[m];   // GELU then folded bn2
      }
    }
    BF16Frag B2;
    pack_pair(hv, h, B2);                       // D-layout -> B-fragment, in registers
    // GEMM-2 partial over this K-block, double-buffered A-fragments
    BF16Frag Acur = load_afrag(w2 + (size_t)n * HIDLK + kb * 32, h);
#pragma unroll
    for (int mt = 0; mt < 8; ++mt) {
      BF16Frag Anext = load_afrag(w2 + (size_t)((((mt + 1) & 7) * 16) + n) * HIDLK + kb * 32, h);
      oacc[mt] = __builtin_amdgcn_wmma_f32_16x16x32_bf16(false, Acur.v, false, B2.v,
                                                         (short)0, oacc[mt], false, false);
      Acur = Anext;
    }
  }

  u16* eob = eo + (size_t)b * CCH * HWSZ + hw;
#pragma unroll
  for (int mt = 0; mt < 8; ++mt)
#pragma unroll
    for (int v = 0; v < 8; ++v) {
      int m = mt * 16 + v + 8 * h;
      eob[(size_t)m * HWSZ] = (u16)f2bf1(oacc[mt][v] * s3[m] + t3[m]);  // folded bn3
    }
}

// ---------------- Std expert fused MLP: eo = W3 @ relu(bn(W2 @ (W1 @ x + b1) + b2)) + b3 ----
__global__ __launch_bounds__(256) void k_std_mlp(const float* __restrict__ xg,  // [B][C][HW] f32
                                                 const u16* __restrict__ w1, const float* __restrict__ b1,
                                                 const u16* __restrict__ w2, const float* __restrict__ b2,
                                                 const u16* __restrict__ w3, const float* __restrict__ b3,
                                                 const float* __restrict__ sb, const float* __restrict__ tb2,
                                                 u16* __restrict__ eo) {
  const int lane = threadIdx.x & 31;
  const int wave = threadIdx.x >> 5;
  const int h = lane >> 4, n = lane & 15;
  const size_t pix0 = (size_t)blockIdx.x * 128 + wave * 16;
  const int b = (int)(pix0 / HWSZ);
  const int hw = (int)(pix0 % HWSZ) + n;
  const float* xb = xg + (size_t)b * CCH * HWSZ + hw;

  BF16Frag B1[4];
#pragma unroll
  for (int kk = 0; kk < 4; ++kk) {
#pragma unroll
    for (int v = 0; v < 8; ++v) {
      int k = kk * 32 + h * 16 + 2 * v;
      B1[kk].u[v] = f2bf_pk(xb[(size_t)k * HWSZ], xb[(size_t)(k + 1) * HWSZ]);
    }
  }

  // Phase 1: H1 = W1 @ x + b1, kept entirely as register B-fragments (256 wide).
  BF16Frag H1B[8];
#pragma unroll 1
  for (int kb = 0; kb < 8; ++kb) {
    float hv[2][8];
#pragma unroll
    for (int ti = 0; ti < 2; ++ti) {
      v8f acc = (v8f){0, 0, 0, 0, 0, 0, 0, 0};
      const u16* wrow = w1 + (size_t)(kb * 32 + ti * 16 + n) * CCH;
      BF16Frag A[4];
#pragma unroll
      for (int kk = 0; kk < 4; ++kk) A[kk] = load_afrag(wrow + kk * 32, h);
#pragma unroll
      for (int kk = 0; kk < 4; ++kk)
        acc = __builtin_amdgcn_wmma_f32_16x16x32_bf16(false, A[kk].v, false, B1[kk].v,
                                                      (short)0, acc, false, false);
#pragma unroll
      for (int v = 0; v < 8; ++v)
        hv[ti][v] = acc[v] + b1[kb * 32 + ti * 16 + v + 8 * h];
    }
    pack_pair(hv, h, H1B[kb]);
  }

  v8f oacc[8];
#pragma unroll
  for (int i = 0; i < 8; ++i) oacc[i] = (v8f){0, 0, 0, 0, 0, 0, 0, 0};

  // Phase 2+3: stream H2 = relu(bn(W2 @ H1 + b2)) in 32-row blocks into GEMM-3.
#pragma unroll 1
  for (int kb = 0; kb < 8; ++kb) {
    float hv[2][8];
#pragma unroll
    for (int ti = 0; ti < 2; ++ti) {
      v8f acc = (v8f){0, 0, 0, 0, 0, 0, 0, 0};
      const u16* wrow = w2 + (size_t)(kb * 32 + ti * 16 + n) * HIDST;
      // double-buffer A across the 8-deep K chain
      BF16Frag Acur = load_afrag(wrow, h);
#pragma unroll
      for (int kk = 0; kk < 8; ++kk) {
        BF16Frag Anext = load_afrag(wrow + ((kk + 1) & 7) * 32, h);
        acc = __builtin_amdgcn_wmma_f32_16x16x32_bf16(false, Acur.v, false, H1B[kk].v,
                                                      (short)0, acc, false, false);
        Acur = Anext;
      }
#pragma unroll
      for (int v = 0; v < 8; ++v) {
        int m = kb * 32 + ti * 16 + v + 8 * h;
        float y = acc[v] + b2[m];
        y = y * sb[m] + tb2[m];                 // folded BN
        hv[ti][v] = fmaxf(y, 0.f);              // ReLU
      }
    }
    BF16Frag B3;
    pack_pair(hv, h, B3);
    BF16Frag Acur = load_afrag(w3 + (size_t)n * HIDST + kb * 32, h);
#pragma unroll
    for (int mt = 0; mt < 8; ++mt) {
      BF16Frag Anext = load_afrag(w3 + (size_t)((((mt + 1) & 7) * 16) + n) * HIDST + kb * 32, h);
      oacc[mt] = __builtin_amdgcn_wmma_f32_16x16x32_bf16(false, Acur.v, false, B3.v,
                                                         (short)0, oacc[mt], false, false);
      Acur = Anext;
    }
  }

  u16* eob = eo + (size_t)b * CCH * HWSZ + hw;
#pragma unroll
  for (int mt = 0; mt < 8; ++mt)
#pragma unroll
    for (int v = 0; v < 8; ++v) {
      int m = mt * 16 + v + 8 * h;
      eob[(size_t)m * HWSZ] = (u16)f2bf1(oacc[mt][v] + b3[m]);
    }
}

// ---------------- weighted combine: out[g] = sum_e w[g,b,e] * eo[e] ----------------
__global__ __launch_bounds__(256) void k_combine(const u16* __restrict__ eo,
                                                 const float* __restrict__ wc,
                                                 float* __restrict__ out) {
  size_t i2 = (size_t)blockIdx.x * 512 + (size_t)threadIdx.x * 2;   // element pair
  if (i2 >= PLANE) return;
  // b is block-uniform (512 | C*HW), so the 16 gate weights become scalar loads
  int b = (int)(((size_t)blockIdx.x * 512) / ((size_t)CCH * HWSZ));
  u32 p0 = *(const u32*)(eo + i2);
  u32 p1 = *(const u32*)(eo + PLANE + i2);
  u32 p2 = *(const u32*)(eo + 2 * PLANE + i2);
  u32 p3 = *(const u32*)(eo + 3 * PLANE + i2);
  float e0l = bf2f((u16)p0), e0h = bf2f((u16)(p0 >> 16));
  float e1l = bf2f((u16)p1), e1h = bf2f((u16)(p1 >> 16));
  float e2l = bf2f((u16)p2), e2h = bf2f((u16)(p2 >> 16));
  float e3l = bf2f((u16)p3), e3h = bf2f((u16)(p3 >> 16));
#pragma unroll
  for (int g = 0; g < 4; ++g) {
    const float* w = wc + (size_t)(g * BB + b) * EE;
    float w0 = w[0], w1 = w[1], w2 = w[2], w3 = w[3];
    float2 r;
    r.x = w0 * e0l + w1 * e1l + w2 * e2l + w3 * e3l;
    r.y = w0 * e0h + w1 * e1h + w2 * e2h + w3 * e3h;
    *(float2*)(out + (size_t)g * PLANE + i2) = r;
  }
}

// ---------------- launcher ----------------
extern "C" void kernel_launch(void* const* d_in, const int* in_sizes, int n_in,
                              void* d_out, int out_size, void* d_ws, size_t ws_size,
                              hipStream_t stream) {
  (void)in_sizes; (void)n_in; (void)out_size; (void)ws_size;
  const float* x     = (const float*)d_in[0];
  const float* gates = (const float*)d_in[1];
  // d_in flattening: x, gates, 3x (dw_w,pw1_w,pw2_w, bn1 g/b/m/v, bn2 g/b/m/v, bn3 g/b/m/v), std 10 arrays
  const int lk_base[3] = {2, 17, 32};
  const int lk_slot[3] = {0, 1, 3};             // hybrid order [LK0, LK1, Std, LK2]
  const int sbase = 47;                          // w1,b1,w2,b2,w3,b3,bn_g,bn_b,bn_m,bn_v

  // ---- workspace carve-out (~95 MB) ----
  char* wsb = (char*)d_ws;
  size_t off = 0;
  auto take = [&](size_t bytes) -> char* {
    char* p = wsb + off;
    off += (bytes + 255) & ~(size_t)255;
    return p;
  };
  float* gap = (float*)take(BB * CCH * sizeof(float));
  float* wc  = (float*)take(4 * BB * EE * sizeof(float));
  float *lks1[3], *lkt1[3], *lks2[3], *lkt2[3], *lks3[3], *lkt3[3];
  for (int e = 0; e < 3; ++e) {
    lks1[e] = (float*)take(CCH * 4);   lkt1[e] = (float*)take(CCH * 4);
    lks2[e] = (float*)take(HIDLK * 4); lkt2[e] = (float*)take(HIDLK * 4);
    lks3[e] = (float*)take(CCH * 4);   lkt3[e] = (float*)take(CCH * 4);
  }
  float* ssb = (float*)take(HIDST * 4);
  float* stb = (float*)take(HIDST * 4);
  u16 *w1bf[3], *w2bf[3];
  for (int e = 0; e < 3; ++e) {
    w1bf[e] = (u16*)take((size_t)HIDLK * CCH * 2);
    w2bf[e] = (u16*)take((size_t)CCH * HIDLK * 2);
  }
  u16* sw1 = (u16*)take((size_t)HIDST * CCH * 2);
  u16* sw2 = (u16*)take((size_t)HIDST * HIDST * 2);
  u16* sw3 = (u16*)take((size_t)CCH * HIDST * 2);
  u16* tbuf = (u16*)take(PLANE * 2);
  u16* eo   = (u16*)take(4 * PLANE * 2);

  // ---- prep: fold BN, convert weights to bf16 ----
  for (int e = 0; e < 3; ++e) {
    int ib = lk_base[e];
    k_fold_bn<<<2, 64, 0, stream>>>((const float*)d_in[ib + 3], (const float*)d_in[ib + 4],
                                    (const float*)d_in[ib + 5], (const float*)d_in[ib + 6],
                                    lks1[e], lkt1[e], CCH);
    k_fold_bn<<<8, 64, 0, stream>>>((const float*)d_in[ib + 7], (const float*)d_in[ib + 8],
                                    (const float*)d_in[ib + 9], (const float*)d_in[ib + 10],
                                    lks2[e], lkt2[e], HIDLK);
    k_fold_bn<<<2, 64, 0, stream>>>((const float*)d_in[ib + 11], (const float*)d_in[ib + 12],
                                    (const float*)d_in[ib + 13], (const float*)d_in[ib + 14],
                                    lks3[e], lkt3[e], CCH);
    k_cvt_bf16<<<256, 256, 0, stream>>>((const float*)d_in[ib + 1], w1bf[e], HIDLK * CCH);
    k_cvt_bf16<<<256, 256, 0, stream>>>((const float*)d_in[ib + 2], w2bf[e], CCH * HIDLK);
  }
  k_fold_bn<<<4, 64, 0, stream>>>((const float*)d_in[sbase + 6], (const float*)d_in[sbase + 7],
                                  (const float*)d_in[sbase + 8], (const float*)d_in[sbase + 9],
                                  ssb, stb, HIDST);
  k_cvt_bf16<<<128, 256, 0, stream>>>((const float*)d_in[sbase + 0], sw1, HIDST * CCH);
  k_cvt_bf16<<<256, 256, 0, stream>>>((const float*)d_in[sbase + 2], sw2, HIDST * HIDST);
  k_cvt_bf16<<<128, 256, 0, stream>>>((const float*)d_in[sbase + 4], sw3, CCH * HIDST);

  // ---- gating ----
  float* out = (float*)d_out;
  k_gap<<<BB * CCH, 256, 0, stream>>>(x, gap);
  k_gate<<<1, 128, 0, stream>>>(gap, gates, wc, out + 4 * PLANE);

  // ---- experts ----
  const int nblk = (int)((BB * (size_t)HWSZ) / 128);   // 576 pixel-tile blocks
  for (int e = 0; e < 3; ++e) {
    int ib = lk_base[e];
    k_dw<<<BB * CCH, 256, 0, stream>>>(x, (const float*)d_in[ib + 0], lks1[e], lkt1[e], tbuf);
    k_lk_mlp<<<nblk, 256, 0, stream>>>(tbuf, w1bf[e], w2bf[e],
                                       lks2[e], lkt2[e], lks3[e], lkt3[e],
                                       eo + (size_t)lk_slot[e] * PLANE);
  }
  k_std_mlp<<<nblk, 256, 0, stream>>>(x, sw1, (const float*)d_in[sbase + 1],
                                      sw2, (const float*)d_in[sbase + 3],
                                      sw3, (const float*)d_in[sbase + 5],
                                      ssb, stb, eo + 2 * PLANE);

  // ---- combine ----
  k_combine<<<(int)((PLANE + 511) / 512), 512 / 2, 0, stream>>>(eo, wc, out);
}
